// Multihead_30245159698495
// MI455X (gfx1250) — compile-verified, matrix operand order checked by reading, not run on hardware
//
#include <hip/hip_runtime.h>
#include <hip/hip_bf16.h>
#include <stdint.h>

// ---------------------------------------------------------------- constants
#define DIMX     1024
#define HEADS    16
#define HEAD_DIM 64
#define SEQ      2048
#define BATCH    4
#define ROWS     (BATCH * SEQ)      // 8192
#define NQKV     (3 * DIMX)         // 3072

typedef __attribute__((ext_vector_type(16))) __bf16       v16bf;
typedef __attribute__((ext_vector_type(8)))  float        v8f;
typedef __attribute__((ext_vector_type(4)))  unsigned int u32x4;
typedef __attribute__((ext_vector_type(8)))  unsigned int u32x8;

// ---------------------------------------------------------------- helpers
__device__ __forceinline__ v8f wmma_bf16(v16bf a, v16bf b, v8f c) {
    // D = A(16x32 bf16) * B(32x16 bf16) + C(16x16 f32)
    return __builtin_amdgcn_wmma_f32_16x16x32_bf16(
        /*neg_a=*/false, a, /*neg_b=*/false, b,
        /*c_mod=*/(short)0, c, /*reuse_a=*/false, /*reuse_b=*/false);
}

// Load a 16x32 bf16 A-fragment (or a 32x16 B-fragment from a [N,K] row-major
// Bt matrix, which has the identical per-lane layout with N in place of M).
// ISA layout: lane = half*16 + r ; this lane holds row r, cols
//   [half*8 .. half*8+7] and [16+half*8 .. 16+half*8+7]  (two 16B chunks).
__device__ __forceinline__ v16bf load_frag(const __bf16* p, int stride, int row) {
    const int lane = threadIdx.x & 31;
    const int half = lane >> 4;
    const int r    = lane & 15;
    const __bf16* q = p + (size_t)(row + r) * stride + half * 8;
    u32x4 lo = *(const u32x4*)(q);
    u32x4 hi = *(const u32x4*)(q + 16);
    u32x8 t;
    t[0] = lo[0]; t[1] = lo[1]; t[2] = lo[2]; t[3] = lo[3];
    t[4] = hi[0]; t[5] = hi[1]; t[6] = hi[2]; t[7] = hi[3];
    return __builtin_bit_cast(v16bf, t);
}

// CDNA5 async global->LDS copy (ASYNCcnt-tracked, no VGPR round trip).
// lds_off = byte offset within the workgroup's LDS allocation (low 32 bits of
// a generic pointer to __shared__), gaddr = 64-bit global address.
__device__ __forceinline__ void async_copy16(uint32_t lds_off, uint64_t gaddr) {
    asm volatile("global_load_async_to_lds_b128 %0, %1, off"
                 :: "v"(lds_off), "v"(gaddr) : "memory");
}
#define WAIT_ASYNC_LE(n) asm volatile("s_wait_asynccnt " #n ::: "memory")

// ---------------------------------------------------------------- kernel 1
// f32 -> bf16 with transpose:  out[n*rows_in + k] = (bf16) in[k*cols_in + n]
__global__ void __launch_bounds__(256)
cvt_transpose_kernel(const float* __restrict__ in, __bf16* __restrict__ out,
                     int rows_in, int cols_in) {
    int t = blockIdx.x * 256 + threadIdx.x;
    int n = t / rows_in;
    int k = t - n * rows_in;
    out[t] = (__bf16)in[(size_t)k * cols_in + n];
}

// ---------------------------------------------------------------- kernel 2
// Pre-norm LayerNorm, f32 in -> bf16 out. One 256-thread block per row.
__global__ void __launch_bounds__(256)
layernorm_kernel(const float* __restrict__ x, const float* __restrict__ gamma,
                 const float* __restrict__ beta, __bf16* __restrict__ xn) {
    const int row = blockIdx.x;
    const int tid = threadIdx.x;
    const float* xr = x + (size_t)row * DIMX;
    float v[4], s = 0.f, s2 = 0.f;
#pragma unroll
    for (int i = 0; i < 4; ++i) {
        v[i] = xr[tid + i * 256];
        s += v[i]; s2 += v[i] * v[i];
    }
#pragma unroll
    for (int off = 16; off; off >>= 1) {
        s  += __shfl_xor(s,  off);
        s2 += __shfl_xor(s2, off);
    }
    __shared__ float red[2][8];
    const int wave = tid >> 5, lane = tid & 31;
    if (lane == 0) { red[0][wave] = s; red[1][wave] = s2; }
    __syncthreads();
    s = 0.f; s2 = 0.f;
#pragma unroll
    for (int i = 0; i < 8; ++i) { s += red[0][i]; s2 += red[1][i]; }
    const float mu   = s * (1.f / DIMX);
    const float var  = s2 * (1.f / DIMX) - mu * mu;
    const float rinv = rsqrtf(var + 1e-5f);
    __bf16* orow = xn + (size_t)row * DIMX;
#pragma unroll
    for (int i = 0; i < 4; ++i) {
        const int c = tid + i * 256;
        orow[c] = (__bf16)((v[i] - mu) * rinv * gamma[c] + beta[c]);
    }
}

// ---------------------------------------------------------------- GEMM tiles
#define BM   128
#define BN   128
#define BK   32
#define NIT  (DIMX / BK)            // 32 K-steps
#define LDT  40                     // padded LDS row stride (bf16): 80B rows
#define TILEB (BM * LDT * 2)        // bytes per LDS tile buffer (10240)

// Shared GEMM body: 128x128x32 block tile, 8 waves, wave tile 32(m) x 64(n),
// async double-buffered global->LDS staging. Each thread stages 4x16B per
// K-step (A rows rr, rr+64; B rows rr, rr+64). Async loads complete in order,
// so waiting ASYNCcnt<=4 after issuing the next tile's 4 guarantees the
// current tile is resident.
#define GEMM_CORE(A_, Bt_, ACC_)                                               \
    __shared__ __align__(16) __bf16 As[2 * BM * LDT];                          \
    __shared__ __align__(16) __bf16 Bs[2 * BN * LDT];                          \
    const int tid  = threadIdx.x;                                              \
    const int wave = tid >> 5;                                                 \
    const int wm   = wave >> 1;   /* 0..3 -> 32-row slice  */                  \
    const int wn   = wave & 1;    /* 0..1 -> 64-col slice  */                  \
    const int block_n = blockIdx.x * BN;                                       \
    const int block_m = blockIdx.y * BM;                                       \
    const v8f zero = {0.f,0.f,0.f,0.f,0.f,0.f,0.f,0.f};                        \
    v8f ACC_[2][4] = {{zero,zero,zero,zero},{zero,zero,zero,zero}};            \
    const int rr = tid >> 2;          /* 0..63    */                           \
    const int cc = (tid & 3) * 8;     /* 0,8,16,24 */                          \
    const uint64_t ga0 = (uint64_t)(A_  + (size_t)(block_m + rr)      * DIMX + cc); \
    const uint64_t ga1 = (uint64_t)(A_  + (size_t)(block_m + rr + 64) * DIMX + cc); \
    const uint64_t gb0 = (uint64_t)(Bt_ + (size_t)(block_n + rr)      * DIMX + cc); \
    const uint64_t gb1 = (uint64_t)(Bt_ + (size_t)(block_n + rr + 64) * DIMX + cc); \
    const uint32_t la0 = (uint32_t)(uintptr_t)(As + rr * LDT + cc);            \
    const uint32_t la1 = (uint32_t)(uintptr_t)(As + (rr + 64) * LDT + cc);     \
    const uint32_t lb0 = (uint32_t)(uintptr_t)(Bs + rr * LDT + cc);            \
    const uint32_t lb1 = (uint32_t)(uintptr_t)(Bs + (rr + 64) * LDT + cc);     \
    /* prologue: stage tile 0 into buffer 0 */                                 \
    async_copy16(la0, ga0); async_copy16(la1, ga1);                            \
    async_copy16(lb0, gb0); async_copy16(lb1, gb1);                            \
    for (int it = 0; it < NIT; ++it) {                                         \
        const int buf = it & 1;                                                \
        if (it + 1 < NIT) {                                                    \
            const uint64_t kb = (uint64_t)(it + 1) * (BK * 2); /* bytes */     \
            const uint32_t lo = (buf ^ 1) * TILEB;                             \
            async_copy16(la0 + lo, ga0 + kb); async_copy16(la1 + lo, ga1 + kb);\
            async_copy16(lb0 + lo, gb0 + kb); async_copy16(lb1 + lo, gb1 + kb);\
            WAIT_ASYNC_LE(4);                                                  \
        } else {                                                               \
            WAIT_ASYNC_LE(0);                                                  \
        }                                                                      \
        __syncthreads();              /* tile `it` visible to all waves */     \
        const __bf16* Ab = As + buf * (BM * LDT);                              \
        const __bf16* Bb = Bs + buf * (BN * LDT);                              \
        v16bf fa0 = load_frag(Ab, LDT, wm * 32);                               \
        v16bf fa1 = load_frag(Ab, LDT, wm * 32 + 16);                          \
        v16bf fb0 = load_frag(Bb, LDT, wn * 64);                               \
        v16bf fb1 = load_frag(Bb, LDT, wn * 64 + 16);                          \
        v16bf fb2 = load_frag(Bb, LDT, wn * 64 + 32);                          \
        v16bf fb3 = load_frag(Bb, LDT, wn * 64 + 48);                          \
        ACC_[0][0] = wmma_bf16(fa0, fb0, ACC_[0][0]);                          \
        ACC_[0][1] = wmma_bf16(fa0, fb1, ACC_[0][1]);                          \
        ACC_[0][2] = wmma_bf16(fa0, fb2, ACC_[0][2]);                          \
        ACC_[0][3] = wmma_bf16(fa0, fb3, ACC_[0][3]);                          \
        ACC_[1][0] = wmma_bf16(fa1, fb0, ACC_[1][0]);                          \
        ACC_[1][1] = wmma_bf16(fa1, fb1, ACC_[1][1]);                          \
        ACC_[1][2] = wmma_bf16(fa1, fb2, ACC_[1][2]);                          \
        ACC_[1][3] = wmma_bf16(fa1, fb3, ACC_[1][3]);                          \
        __syncthreads();              /* reads done before next async writes */\
    }

// ---------------------------------------------------------------- kernel 3
// QKV GEMM: xn[8192,1024] (bf16) x wqkvT[3072,1024] (bf16, pre-transposed)
// epilogue scatters Q,K as [b,h,l,d] and V transposed as [b,h,d,l] in bf16.
__global__ void __launch_bounds__(256)
qkv_gemm_kernel(const __bf16* __restrict__ A, const __bf16* __restrict__ Bt,
                const float* __restrict__ b_qkv,
                __bf16* __restrict__ Qb, __bf16* __restrict__ Kb,
                __bf16* __restrict__ VT) {
    GEMM_CORE(A, Bt, acc)

    const int lane = tid & 31;
    const int half = lane >> 4;
    const int ln   = lane & 15;
#pragma unroll
    for (int i = 0; i < 2; ++i) {
#pragma unroll
        for (int j = 0; j < 4; ++j) {
            const int n     = block_n + wn * 64 + j * 16 + ln;
            const int sel   = n >> 10;      // 0:Q 1:K 2:V
            const int inner = n & 1023;
            const int h     = inner >> 6;
            const int dch   = inner & 63;
            const float bias = b_qkv[n];
            const int m_base = block_m + wm * 32 + i * 16 + 8 * half;
#pragma unroll
            for (int r = 0; r < 8; ++r) {
                const int m     = m_base + r;
                const int batch = m >> 11;       // /SEQ
                const int l     = m & 2047;
                const int bh    = batch * HEADS + h;
                const __bf16 bv = (__bf16)(acc[i][j][r] + bias);
                if (sel == 0)      Qb[((size_t)bh * SEQ + l) * HEAD_DIM + dch] = bv;
                else if (sel == 1) Kb[((size_t)bh * SEQ + l) * HEAD_DIM + dch] = bv;
                else               VT[((size_t)bh * HEAD_DIM + dch) * SEQ + l] = bv;
            }
        }
    }
}

// ---------------------------------------------------------------- kernel 4
// Flash attention. One wave owns a 16-row Q tile; 8 waves/block -> 128 q rows.
// Keys streamed in chunks of 32; online softmax; P bounced through per-wave LDS.
__global__ void __launch_bounds__(256)
attention_kernel(const __bf16* __restrict__ Qb, const __bf16* __restrict__ Kb,
                 const __bf16* __restrict__ VT, __bf16* __restrict__ attout) {
    const int tid  = threadIdx.x;
    const int wave = tid >> 5;
    const int lane = tid & 31;
    const int half = lane >> 4;
    const int ln   = lane & 15;
    const int bh    = blockIdx.y;                  // 0..63 = batch*16+head
    const int q0    = blockIdx.x * 128 + wave * 16;
    const int batch = bh >> 4;
    const int h     = bh & 15;

    __shared__ __align__(16) __bf16 Ps[8][16 * 32]; // per-wave P bounce buffer
    __bf16* myP = Ps[wave];

    const __bf16* Qbase = Qb + ((size_t)bh * SEQ + q0) * HEAD_DIM;
    const __bf16* Kbase = Kb + (size_t)bh * SEQ * HEAD_DIM;
    const __bf16* Vbase = VT + (size_t)bh * HEAD_DIM * SEQ;

    const v16bf qa0 = load_frag(Qbase,      HEAD_DIM, 0);   // k = d 0..31
    const v16bf qa1 = load_frag(Qbase + 32, HEAD_DIM, 0);   // k = d 32..63

    const v8f zero = {0.f,0.f,0.f,0.f,0.f,0.f,0.f,0.f};
    v8f o[4] = {zero, zero, zero, zero};
    float m[8], l[8];
#pragma unroll
    for (int r = 0; r < 8; ++r) { m[r] = -1e30f; l[r] = 0.f; }

    const float scale = 0.125f;  // HEAD_DIM^-0.5

    for (int kc = 0; kc < SEQ; kc += 32) {
        // ---- scores S[16 q x 32 keys] = Q * K^T
        v8f s0 = zero, s1 = zero;
        s0 = wmma_bf16(qa0, load_frag(Kbase + (size_t)kc * HEAD_DIM,            HEAD_DIM, 0), s0);
        s0 = wmma_bf16(qa1, load_frag(Kbase + (size_t)kc * HEAD_DIM + 32,       HEAD_DIM, 0), s0);
        s1 = wmma_bf16(qa0, load_frag(Kbase + (size_t)(kc + 16) * HEAD_DIM,     HEAD_DIM, 0), s1);
        s1 = wmma_bf16(qa1, load_frag(Kbase + (size_t)(kc + 16) * HEAD_DIM + 32,HEAD_DIM, 0), s1);

        // ---- online softmax (row M = r + 8*half lives across a 16-lane group)
#pragma unroll
        for (int r = 0; r < 8; ++r) {
            const float a = s0[r] * scale;
            const float b = s1[r] * scale;
            float cm = fmaxf(a, b);
            cm = fmaxf(cm, __shfl_xor(cm, 1));
            cm = fmaxf(cm, __shfl_xor(cm, 2));
            cm = fmaxf(cm, __shfl_xor(cm, 4));
            cm = fmaxf(cm, __shfl_xor(cm, 8));
            const float mn   = fmaxf(m[r], cm);
            const float corr = __expf(m[r] - mn);
            const float p0   = __expf(a - mn);
            const float p1   = __expf(b - mn);
            float ps = p0 + p1;
            ps += __shfl_xor(ps, 1);
            ps += __shfl_xor(ps, 2);
            ps += __shfl_xor(ps, 4);
            ps += __shfl_xor(ps, 8);
            l[r] = l[r] * corr + ps;
            m[r] = mn;
            const int row = r + 8 * half;          // C-layout -> LDS (A-layout src)
            myP[row * 32 + ln]      = (__bf16)p0;
            myP[row * 32 + 16 + ln] = (__bf16)p1;
#pragma unroll
            for (int j = 0; j < 4; ++j) o[j][r] *= corr;
        }
        // LDS RAW inside the same wave: DS ops are in-order, but fence the
        // compiler + outstanding ds stores before re-reading as an A-fragment.
        asm volatile("s_wait_dscnt 0" ::: "memory");

        // ---- O += P[16x32] * V[32 keys x 64 d]   (V stored transposed [d,l])
        const v16bf pa = load_frag(myP, 32, 0);
#pragma unroll
        for (int j = 0; j < 4; ++j)
            o[j] = wmma_bf16(pa, load_frag(Vbase + (size_t)(j * 16) * SEQ + kc, SEQ, 0), o[j]);
    }

    // ---- finalize: O /= l, write [b, l, h*64 + d] as bf16 for out-proj GEMM
#pragma unroll
    for (int r = 0; r < 8; ++r) {
        const float inv = 1.f / l[r];
        const size_t mrow = (size_t)batch * SEQ + q0 + r + 8 * half;
#pragma unroll
        for (int j = 0; j < 4; ++j)
            attout[mrow * DIMX + h * HEAD_DIM + j * 16 + ln] = (__bf16)(o[j][r] * inv);
    }
}

// ---------------------------------------------------------------- kernel 5
// Out projection: attout[8192,1024] x woutT[1024,1024] + b_out -> f32 d_out
__global__ void __launch_bounds__(256)
out_gemm_kernel(const __bf16* __restrict__ A, const __bf16* __restrict__ Bt,
                const float* __restrict__ b_out, float* __restrict__ out) {
    GEMM_CORE(A, Bt, acc)

    const int lane = tid & 31;
    const int half = lane >> 4;
    const int ln   = lane & 15;
#pragma unroll
    for (int i = 0; i < 2; ++i) {
#pragma unroll
        for (int j = 0; j < 4; ++j) {
            const int n = block_n + wn * 64 + j * 16 + ln;
            const float bias = b_out[n];
            const int m_base = block_m + wm * 32 + i * 16 + 8 * half;
#pragma unroll
            for (int r = 0; r < 8; ++r)
                out[(size_t)(m_base + r) * DIMX + n] = acc[i][j][r] + bias;
        }
    }
}

// ---------------------------------------------------------------- launcher
extern "C" void kernel_launch(void* const* d_in, const int* in_sizes, int n_in,
                              void* d_out, int out_size, void* d_ws, size_t ws_size,
                              hipStream_t stream) {
    const float* x      = (const float*)d_in[0];   // [4,2048,1024]
    const float* w_qkv  = (const float*)d_in[1];   // [1024,3072]
    const float* b_qkv  = (const float*)d_in[2];   // [3072]
    const float* w_out  = (const float*)d_in[3];   // [1024,1024]
    const float* b_out  = (const float*)d_in[4];   // [1024]
    const float* gamma  = (const float*)d_in[5];   // [1024]
    const float* beta   = (const float*)d_in[6];   // [1024]
    float* out = (float*)d_out;                    // [4,2048,1024]

    char* ws = (char*)d_ws;                        // 88 MiB used
    __bf16* xn     = (__bf16*)(ws);                          // 16 MiB  [8192,1024]
    __bf16* wqkvT  = (__bf16*)(ws + ((size_t)16 << 20));     //  6 MiB  [3072,1024]
    __bf16* woutT  = (__bf16*)(ws + ((size_t)22 << 20));     //  2 MiB  [1024,1024]
    __bf16* Qb     = (__bf16*)(ws + ((size_t)24 << 20));     // 16 MiB  [b,h,l,d]
    __bf16* Kb     = (__bf16*)(ws + ((size_t)40 << 20));     // 16 MiB  [b,h,l,d]
    __bf16* VT     = (__bf16*)(ws + ((size_t)56 << 20));     // 16 MiB  [b,h,d,l]
    __bf16* attout = (__bf16*)(ws + ((size_t)72 << 20));     // 16 MiB  [8192,1024]

    cvt_transpose_kernel<<<(DIMX * NQKV) / 256, 256, 0, stream>>>(w_qkv, wqkvT, DIMX, NQKV);
    cvt_transpose_kernel<<<(DIMX * DIMX) / 256, 256, 0, stream>>>(w_out, woutT, DIMX, DIMX);
    layernorm_kernel<<<ROWS, 256, 0, stream>>>(x, gamma, beta, xn);
    qkv_gemm_kernel<<<dim3(NQKV / BN, ROWS / BM), 256, 0, stream>>>(xn, wqkvT, b_qkv, Qb, Kb, VT);
    attention_kernel<<<dim3(SEQ / 128, BATCH * HEADS), 256, 0, stream>>>(Qb, Kb, VT, attout);
    out_gemm_kernel<<<dim3(DIMX / BN, ROWS / BM), 256, 0, stream>>>(attout, woutT, b_out, out);
}
